// AttenOpWithKV_80547816669945
// MI455X (gfx1250) — matrix operand order repaired and checked by reading, not compiled, not run on hardware
//
#include <hip/hip_runtime.h>
#include <math.h>

// ---------------- problem constants (from reference setup_inputs) ----------
#define B_     8
#define SQ_    16
#define SKV_   16
#define H_     16
#define DH_    128
#define CACHE_ 8192
#define NW     8      // waves per workgroup
#define TK     32     // keys per tile (WMMA K dim for PV)
#define WS_STRIDE 2080  // per (bh,s) partial: M[16] + L[16] + O[16*128]

typedef __bf16 bf16_t;
typedef __attribute__((ext_vector_type(16))) bf16_t v16bf;
typedef __attribute__((ext_vector_type(8)))  float  v8f;

union BF8 { unsigned u[8]; v16bf v; };

// pack two f32 -> packed bf16 pair (low = a, high = b) using HW converter
static __device__ __forceinline__ unsigned pkbf(float a, float b) {
#if defined(__has_builtin) && __has_builtin(__builtin_amdgcn_cvt_pk_bf16_f32)
    auto pk = __builtin_amdgcn_cvt_pk_bf16_f32(a, b);
    unsigned u;
    __builtin_memcpy(&u, &pk, sizeof(u));
    return u;
#else
    union { bf16_t h[2]; unsigned u; } r;
    r.h[0] = (bf16_t)a; r.h[1] = (bf16_t)b;
    return r.u;
#endif
}
static __device__ __forceinline__ unsigned short f2bf(float a) {
    union { bf16_t h; unsigned short u; } r;
    r.h = (bf16_t)a;
    return r.u;
}
static __device__ __forceinline__ int imin(int a, int b) { return a < b ? a : b; }
static __device__ __forceinline__ int imax(int a, int b) { return a > b ? a : b; }

// row pointer for logical key index: history from cache, tail from new k/v
static __device__ __forceinline__ const float*
key_row(const float* __restrict__ cache, const float* __restrict__ neu,
        int b, int h, int key, int start) {
    if (key < start)
        return cache + (((size_t)b * CACHE_ + key) * H_ + h) * DH_;
    return neu + (((size_t)b * SKV_ + (key - start)) * H_ + h) * DH_;
}

// ---------------------------------------------------------------------------
// Phase 1: flash attention over a KV span; write unnormalized partial
// (M[16], L[16], O[16][128]) to workspace.
// blockIdx.x = bh * S + s
// ---------------------------------------------------------------------------
__global__ void __launch_bounds__(256)
fa_phase1(const float* __restrict__ q, const float* __restrict__ k,
          const float* __restrict__ v, const float* __restrict__ kc,
          const float* __restrict__ vc, const int* __restrict__ sidx,
          float* __restrict__ ws, int S)
{
    __shared__ __align__(16) unsigned short p_lds[NW][SQ_][TK]; // probs bf16
    __shared__ float m_lds[NW][SQ_];
    __shared__ float l_lds[NW][SQ_];
    __shared__ float o_acc[SQ_][DH_];

    const int tid  = threadIdx.x;
    const int wave = tid >> 5;
    const int lane = tid & 31;
    const int half = lane >> 4;       // 0: lanes 0-15, 1: lanes 16-31
    const int l16  = lane & 15;

    const int bh = blockIdx.x / S;
    const int s  = blockIdx.x % S;
    const int b  = bh / H_;
    const int h  = bh % H_;

    const int start  = *sidx;
    const int kv_len = start + SKV_;
    const float scale = rsqrtf((float)DH_);

    // this WG's KV span
    const int span    = (kv_len + S - 1) / S;
    const int k_begin = s * span;
    const int k_end   = imin(kv_len, k_begin + span);
    const int tiles   = (imax(0, k_end - k_begin) + TK - 1) / TK;
    const int n_iter  = (tiles + NW - 1) / NW;

    // zero the shared accumulator
    for (int i = tid; i < SQ_ * DH_; i += 256) ((float*)o_acc)[i] = 0.f;
    __syncthreads();

    // ---- Q as 4 bf16 A-tiles, scale folded in ----
    // A 16-bit 16x32 layout: lane<16 -> row=l16, K={0..7,16..23};
    //                        lane>=16 -> row=l16, K={8..15,24..31}
    v16bf qa[4];
    {
        const float* qrow = q + (((size_t)b * SQ_ + l16) * H_ + h) * DH_;
        #pragma unroll
        for (int c = 0; c < 4; ++c) {
            BF8 t;
            const int d0 = 32 * c + half * 8;
            #pragma unroll
            for (int p = 0; p < 4; ++p) {
                t.u[p]     = pkbf(qrow[d0 + 2*p] * scale,      qrow[d0 + 2*p + 1] * scale);
                t.u[4 + p] = pkbf(qrow[d0 + 16 + 2*p] * scale, qrow[d0 + 16 + 2*p + 1] * scale);
            }
            qa[c] = t.v;
        }
    }

    // running softmax state; lane covers rows r + half*8, r = 0..7
    float m_run[8], l_run[8];
    v8f o_t[8];
    #pragma unroll
    for (int r = 0; r < 8; ++r) { m_run[r] = -1e30f; l_run[r] = 0.f; }
    #pragma unroll
    for (int c = 0; c < 8; ++c) { v8f z = {}; o_t[c] = z; }

    for (int it = 0; it < n_iter; ++it) {
        const int base = k_begin + (wave + it * NW) * TK;

        // prefetch next tile's K/V rows into L2 (one row per lane)
        {
            const int nbase = base + NW * TK;
            if (nbase < k_end) {
                const int pkey = imin(nbase + lane, k_end - 1);
                __builtin_prefetch(key_row(kc, k, b, h, pkey, start), 0, 1);
                __builtin_prefetch(key_row(vc, v, b, h, pkey, start), 0, 1);
            }
        }

        // ---- scores: two 16x16 f32 C tiles (keys base..+15, base+16..+31) --
        v8f s0 = {}, s1 = {};
        #pragma unroll
        for (int g = 0; g < 2; ++g) {
            const int key  = base + g * 16 + l16;
            const int keyc = key < k_end ? key : k_end - 1;
            // B tile (d x key): lane half picks d {0..7,16..23} / {8..15,24..31} of the chunk
            const float* kr = key_row(kc, k, b, h, keyc, start) + half * 8;
            #pragma unroll
            for (int c = 0; c < 4; ++c) {
                BF8 t;
                const float* p4 = kr + 32 * c;
                #pragma unroll
                for (int p = 0; p < 8; ++p)
                    t.u[p] = pkbf(p4[2*p + (p >= 4 ? 8 : 0)], p4[2*p + 1 + (p >= 4 ? 8 : 0)]);
                if (g == 0)
                    s0 = __builtin_amdgcn_wmma_f32_16x16x32_bf16(
                        false, qa[c], false, t.v, (short)0, s0, false, false);
                else
                    s1 = __builtin_amdgcn_wmma_f32_16x16x32_bf16(
                        false, qa[c], false, t.v, (short)0, s1, false, false);
            }
        }

        // mask invalid key columns (column index = l16 per C-tile)
        const float msk0 = (base + l16)      < k_end ? 1.f : 0.f;
        const float msk1 = (base + 16 + l16) < k_end ? 1.f : 0.f;
        if (msk0 == 0.f) {
            #pragma unroll
            for (int r = 0; r < 8; ++r) s0[r] = -1e30f;
        }
        if (msk1 == 0.f) {
            #pragma unroll
            for (int r = 0; r < 8; ++r) s1[r] = -1e30f;
        }

        // ---- online softmax update ----
        float alpha[8];
        #pragma unroll
        for (int r = 0; r < 8; ++r) {
            float mx = fmaxf(s0[r], s1[r]);
            #pragma unroll
            for (int sf = 1; sf < 16; sf <<= 1)
                mx = fmaxf(mx, __shfl_xor(mx, sf, 32));
            const float mnew = fmaxf(m_run[r], mx);
            alpha[r] = __expf(m_run[r] - mnew);
            const float p0 = __expf(s0[r] - mnew) * msk0;
            const float p1 = __expf(s1[r] - mnew) * msk1;
            s0[r] = p0; s1[r] = p1;
            float rs = p0 + p1;
            #pragma unroll
            for (int sf = 1; sf < 16; sf <<= 1)
                rs += __shfl_xor(rs, sf, 32);
            l_run[r] = l_run[r] * alpha[r] + rs;
            m_run[r] = mnew;
        }
        #pragma unroll
        for (int c = 0; c < 8; ++c) {
            #pragma unroll
            for (int r = 0; r < 8; ++r)
                o_t[c][r] *= alpha[r];
        }

        // ---- probs C-layout -> A-layout via per-wave LDS transpose ----
        #pragma unroll
        for (int r = 0; r < 8; ++r) {
            const int row = r + half * 8;
            p_lds[wave][row][l16]      = f2bf(s0[r]);
            p_lds[wave][row][16 + l16] = f2bf(s1[r]);
        }
        // same-wave LDS ops are in-order; compiler inserts s_wait_dscnt for
        // the dependent register uses below.
        v16bf pa;
        {
            const unsigned short* prow = &p_lds[wave][l16][0];
            const uint4 x = *(const uint4*)(prow + half * 8);       // keys h8..h8+7
            const uint4 y = *(const uint4*)(prow + 16 + half * 8);  // keys 16+h8..
            BF8 t;
            t.u[0] = x.x; t.u[1] = x.y; t.u[2] = x.z; t.u[3] = x.w;
            t.u[4] = y.x; t.u[5] = y.y; t.u[6] = y.z; t.u[7] = y.w;
            pa = t.v;
        }

        // ---- O += P (16x32) x V (32x128), 8 column chunks of 16 ----
        #pragma unroll
        for (int c = 0; c < 8; ++c) {
            BF8 t;
            const int dcol = 16 * c + l16;
            #pragma unroll
            for (int p = 0; p < 8; ++p) {
                const int ka = base + half * 8 + 2 * p + (p >= 4 ? 8 : 0);
                const int kca = ka     < k_end ? ka     : k_end - 1;
                const int kcb = ka + 1 < k_end ? ka + 1 : k_end - 1;
                const float va = key_row(vc, v, b, h, kca, start)[dcol];
                const float vb = key_row(vc, v, b, h, kcb, start)[dcol];
                t.u[p] = pkbf(va, vb);
            }
            o_t[c] = __builtin_amdgcn_wmma_f32_16x16x32_bf16(
                false, pa, false, t.v, (short)0, o_t[c], false, false);
        }
    }

    // ---- publish per-wave m/l; merge 8 waves inside the WG ----
    if (l16 == 0) {
        #pragma unroll
        for (int r = 0; r < 8; ++r) {
            m_lds[wave][r + half * 8] = m_run[r];
            l_lds[wave][r + half * 8] = l_run[r];
        }
    }
    __syncthreads();

    #pragma unroll
    for (int r = 0; r < 8; ++r) {
        const int row = r + half * 8;
        float M = -1e30f;
        for (int w = 0; w < NW; ++w) M = fmaxf(M, m_lds[w][row]);
        const float scl = __expf(m_run[r] - M);
        #pragma unroll
        for (int c = 0; c < 8; ++c)
            atomicAdd(&o_acc[row][16 * c + l16], o_t[c][r] * scl);
    }
    __syncthreads();

    // ---- write partial (M, L, unnormalized O) to workspace ----
    float* wp = ws + (size_t)blockIdx.x * WS_STRIDE;
    if (tid < SQ_) {
        float M = -1e30f;
        for (int w = 0; w < NW; ++w) M = fmaxf(M, m_lds[w][tid]);
        float L = 0.f;
        for (int w = 0; w < NW; ++w) L += l_lds[w][tid] * __expf(m_lds[w][tid] - M);
        wp[tid]       = M;
        wp[16 + tid]  = L;
    }
    for (int i = tid; i < SQ_ * DH_; i += 256)
        wp[32 + i] = o_acc[i / DH_][i % DH_];
}

// ---------------------------------------------------------------------------
// Phase 2: merge S partials per (b,h), normalize, write output
// out[b][q][h*DH + d]
// ---------------------------------------------------------------------------
__global__ void __launch_bounds__(256)
fa_phase2(const float* __restrict__ ws, float* __restrict__ out, int S)
{
    const int bh = blockIdx.x;
    const int b = bh / H_, h = bh % H_;
    for (int i = threadIdx.x; i < SQ_ * DH_; i += 256) {
        const int row = i / DH_, d = i % DH_;
        float Mg = -1e30f;
        for (int s = 0; s < S; ++s)
            Mg = fmaxf(Mg, ws[((size_t)bh * S + s) * WS_STRIDE + row]);
        float L = 0.f, O = 0.f;
        for (int s = 0; s < S; ++s) {
            const float* p = ws + ((size_t)bh * S + s) * WS_STRIDE;
            const float w = __expf(p[row] - Mg);
            L += p[16 + row] * w;
            O += p[32 + i] * w;
        }
        out[(((size_t)b * SQ_ + row) * H_ + h) * DH_ + d] = O / L;
    }
}

// ---------------------------------------------------------------------------
extern "C" void kernel_launch(void* const* d_in, const int* in_sizes, int n_in,
                              void* d_out, int out_size, void* d_ws, size_t ws_size,
                              hipStream_t stream)
{
    const float* q  = (const float*)d_in[0];
    const float* k  = (const float*)d_in[1];
    const float* v  = (const float*)d_in[2];
    const float* kc = (const float*)d_in[3];
    const float* vc = (const float*)d_in[4];
    const int* sidx = (const int*)d_in[5];
    float* out = (float*)d_out;
    float* ws  = (float*)d_ws;

    // flash-decoding splits: as many as workspace allows (occupancy: 128*S WGs)
    const size_t per = (size_t)B_ * H_ * WS_STRIDE * sizeof(float);
    int S = 1;
    if      (ws_size >= 8 * per) S = 8;
    else if (ws_size >= 4 * per) S = 4;
    else if (ws_size >= 2 * per) S = 2;

    fa_phase1<<<dim3(B_ * H_ * S), dim3(256), 0, stream>>>(q, k, v, kc, vc, sidx, ws, S);
    fa_phase2<<<dim3(B_ * H_), dim3(256), 0, stream>>>(ws, out, S);
}
